// LastBboxTokenizer_68092411510797
// MI455X (gfx1250) — matrix operand order repaired and checked by reading, not compiled, not run on hardware
//
#include <hip/hip_runtime.h>
#include <hip/hip_bf16.h>

typedef __attribute__((ext_vector_type(16))) _Float16 v16h;
typedef __attribute__((ext_vector_type(8)))  _Float16 v8h;
typedef __attribute__((ext_vector_type(8)))  float    v8f;

#define ROWS   (512 * 512)   // B*N
#define HID    256
#define LDB    264           // padded LDS row stride (halves): kills bank conflicts
#define BN_EPS 1e-5f

// ---------------------------------------------------------------------------
// K0: zero the stats area (sum1, sq1, sc1, sh1, sum2, sq2, sc2, sh2, cnt)
// ---------------------------------------------------------------------------
__global__ void k_zero_stats(float* __restrict__ stats, int n) {
    int i = blockIdx.x * blockDim.x + threadIdx.x;
    if (i < n) stats[i] = 0.0f;
}

// ---------------------------------------------------------------------------
// Kprep: W [K=256][N=256] f32 row-major  ->  WT [N][K] f16 row-major
// ---------------------------------------------------------------------------
__global__ void k_prep_weight(const float* __restrict__ W, _Float16* __restrict__ WT) {
    int n = blockIdx.x;
    int k = threadIdx.x;
    WT[n * HID + k] = (_Float16)W[k * HID + n];
}

// ---------------------------------------------------------------------------
// K1: layer 1 (K=4 -> pure VALU). h1pre = x @ W1 + b1 (f16 out),
//     masked per-column sum / sumsq, and mask count.
// ---------------------------------------------------------------------------
__global__ __launch_bounds__(256)
void k1_layer1(const float* __restrict__ bbox, const int* __restrict__ mask,
               const float* __restrict__ W1, const float* __restrict__ b1,
               _Float16* __restrict__ H1, float* __restrict__ sum1,
               float* __restrict__ sq1, float* __restrict__ cnt) {
    __shared__ float xs[32 * 4];
    __shared__ float mks[32];
    const int t = threadIdx.x;
    const long rb = (long)blockIdx.x * 32;

    if (t < 128) xs[t] = bbox[(rb + (t >> 2)) * 16 + (t & 3)];
    if (t < 32)  mks[t] = mask[rb + t] ? 1.0f : 0.0f;
    __syncthreads();

    const float w0 = W1[t], w1 = W1[HID + t], w2 = W1[2 * HID + t], w3 = W1[3 * HID + t];
    const float bj = b1[t];
    float s = 0.0f, q = 0.0f;
#pragma unroll 4
    for (int r = 0; r < 32; ++r) {
        float h = bj + xs[r * 4] * w0 + xs[r * 4 + 1] * w1
                     + xs[r * 4 + 2] * w2 + xs[r * 4 + 3] * w3;
        H1[(size_t)(rb + r) * HID + t] = (_Float16)h;
        const float m = mks[r];
        s += m * h;
        q += m * h * h;
    }
    atomicAdd(&sum1[t], s);
    atomicAdd(&sq1[t], q);
    if (t == 0) {
        float c = 0.0f;
        for (int r = 0; r < 32; ++r) c += mks[r];
        atomicAdd(cnt, c);
    }
}

// ---------------------------------------------------------------------------
// K2/K4: fold masked BN stats into per-feature scale/shift
// ---------------------------------------------------------------------------
__global__ void k_finalize_bn(const float* __restrict__ sum, const float* __restrict__ sq,
                              const float* __restrict__ gamma, const float* __restrict__ beta,
                              const float* __restrict__ cnt,
                              float* __restrict__ scale, float* __restrict__ shift) {
    int j = threadIdx.x;
    float c   = fmaxf(cnt[0], 1.0f);
    float mu  = sum[j] / c;
    float var = sq[j] / c - mu * mu;
    float s   = gamma[j] * rsqrtf(var + BN_EPS);
    scale[j] = s;
    shift[j] = beta[j] - mu * s;
}

// ---------------------------------------------------------------------------
// Shared device helpers for the WMMA GEMM kernels
// ---------------------------------------------------------------------------
__device__ __forceinline__ void stage_B_to_lds(_Float16* sB, const _Float16* BT, int t) {
    // copy 256x256 halves (global, row stride 256) -> LDS (row stride LDB)
#pragma unroll
    for (int it = 0; it < 32; ++it) {
        int idx = it * 2048 + t * 8;          // 256 thr * 8 halves
        int row = idx >> 8;
        int col = idx & 255;
        *(v8h*)&sB[row * LDB + col] = *(const v8h*)(BT + row * HID + col);
    }
}

__device__ __forceinline__ v16h bn_relu_pack(v8h r0, v8h r1, const float* sSc,
                                             const float* sSh, int k2) {
    union { v16h v; v8h h[2]; } a;
#pragma unroll
    for (int i = 0; i < 8; ++i) {
        float v0 = (float)r0[i] * sSc[k2 + i]      + sSh[k2 + i];
        float v1 = (float)r1[i] * sSc[k2 + 16 + i] + sSh[k2 + 16 + i];
        a.h[0][i] = (_Float16)fmaxf(v0, 0.0f);
        a.h[1][i] = (_Float16)fmaxf(v1, 0.0f);
    }
    return a.v;
}

// ---------------------------------------------------------------------------
// K3: layer 2 GEMM via v_wmma_f32_16x16x32_f16.
//   B staged in LDS (padded rows); the wave's ENTIRE A row (8 k-steps) is
//   issued as one clause of global_load_b128 up front so HBM latency is
//   hidden behind ~7 k-steps of WMMA work (loads retire in order ->
//   staggered s_wait_loadcnt).
//   A = relu(BN1(h1pre)); epilogue adds b2, writes h2pre f16, masked stats.
// One wave = 16 rows x 256 cols (16 accumulator tiles). 8 waves/block.
// ---------------------------------------------------------------------------
__global__ __launch_bounds__(256)
void k3_gemm_bn_stats(const _Float16* __restrict__ Apre, const _Float16* __restrict__ BT,
                      const float* __restrict__ scale, const float* __restrict__ shift,
                      const float* __restrict__ bias, const int* __restrict__ mask,
                      _Float16* __restrict__ Cpre,
                      float* __restrict__ gsum, float* __restrict__ gsq) {
    extern __shared__ _Float16 sB[];                    // 256 * LDB halves = 132 KB
    __shared__ float sSc[HID], sSh[HID], sSum[HID], sSq[HID];
    const int t = threadIdx.x;
    sSc[t] = scale[t]; sSh[t] = shift[t];
    sSum[t] = 0.0f;    sSq[t] = 0.0f;
    stage_B_to_lds(sB, BT, t);
    __syncthreads();

    const int wave = t >> 5, lane = t & 31;
    const int l15 = lane & 15, hi = lane >> 4;
    const long mbase = ((long)blockIdx.x * 8 + wave) * 16;

    v8f acc[16];
#pragma unroll
    for (int n = 0; n < 16; ++n) acc[n] = (v8f)(0.0f);

    const size_t aRow = (size_t)(mbase + l15) * HID;
    const _Float16* bBase = sB + l15 * LDB + hi * 8;    // + n*16*LDB + kb

    // preload the whole A row: 16 global_load_b128 in flight
    v8h ra0[8], ra1[8];
#pragma unroll
    for (int kk = 0; kk < 8; ++kk) {
        ra0[kk] = *(const v8h*)(Apre + aRow + kk * 32 + hi * 8);
        ra1[kk] = *(const v8h*)(Apre + aRow + kk * 32 + hi * 8 + 16);
    }

#pragma unroll
    for (int kk = 0; kk < 8; ++kk) {
        const int kb = kk * 32;
        v16h a = bn_relu_pack(ra0[kk], ra1[kk], sSc, sSh, kb + hi * 8);
#pragma unroll
        for (int n = 0; n < 16; ++n) {
            v16h b = __builtin_shufflevector(*(const v8h*)(bBase + n * 16 * LDB + kb), *(const v8h*)(bBase + n * 16 * LDB + kb + 16), 0, 1, 2, 3, 4, 5, 6, 7, 8, 9, 10, 11, 12, 13, 14, 15);
            acc[n] = __builtin_amdgcn_wmma_f32_16x16x32_f16(
                false, a, false, b, (short)0, acc[n], false, false);
        }
    }

    // Epilogue: row = mbase + hi*8 + r, col = n*16 + l15
    const long rowb = mbase + hi * 8;
    float mk[8];
#pragma unroll
    for (int r = 0; r < 8; ++r) mk[r] = mask[rowb + r] ? 1.0f : 0.0f;

#pragma unroll
    for (int n = 0; n < 16; ++n) {
        const int col = n * 16 + l15;
        const float bb = bias[col];
        float ls = 0.0f, lq = 0.0f;
#pragma unroll
        for (int r = 0; r < 8; ++r) {
            float v = acc[n][r] + bb;
            Cpre[(size_t)(rowb + r) * HID + col] = (_Float16)v;
            ls += mk[r] * v;
            lq += mk[r] * v * v;
        }
        atomicAdd(&sSum[col], ls);   // ds_add_f32
        atomicAdd(&sSq[col],  lq);
    }
    __syncthreads();
    atomicAdd(&gsum[t], sSum[t]);
    atomicAdd(&gsq[t],  sSq[t]);
}

// ---------------------------------------------------------------------------
// K5: layer 3 GEMM.  A = relu(BN2(h2pre)), out = (A@W3 + b3) * mask  (f32)
// ---------------------------------------------------------------------------
__global__ __launch_bounds__(256)
void k5_gemm_out(const _Float16* __restrict__ Apre, const _Float16* __restrict__ BT,
                 const float* __restrict__ scale, const float* __restrict__ shift,
                 const float* __restrict__ bias, const int* __restrict__ mask,
                 float* __restrict__ out) {
    extern __shared__ _Float16 sB[];                    // 132 KB
    __shared__ float sSc[HID], sSh[HID];
    const int t = threadIdx.x;
    sSc[t] = scale[t]; sSh[t] = shift[t];
    stage_B_to_lds(sB, BT, t);
    __syncthreads();

    const int wave = t >> 5, lane = t & 31;
    const int l15 = lane & 15, hi = lane >> 4;
    const long mbase = ((long)blockIdx.x * 8 + wave) * 16;

    v8f acc[16];
#pragma unroll
    for (int n = 0; n < 16; ++n) acc[n] = (v8f)(0.0f);

    const size_t aRow = (size_t)(mbase + l15) * HID;
    const _Float16* bBase = sB + l15 * LDB + hi * 8;

    v8h ra0[8], ra1[8];
#pragma unroll
    for (int kk = 0; kk < 8; ++kk) {
        ra0[kk] = *(const v8h*)(Apre + aRow + kk * 32 + hi * 8);
        ra1[kk] = *(const v8h*)(Apre + aRow + kk * 32 + hi * 8 + 16);
    }

#pragma unroll
    for (int kk = 0; kk < 8; ++kk) {
        const int kb = kk * 32;
        v16h a = bn_relu_pack(ra0[kk], ra1[kk], sSc, sSh, kb + hi * 8);
#pragma unroll
        for (int n = 0; n < 16; ++n) {
            v16h b = __builtin_shufflevector(*(const v8h*)(bBase + n * 16 * LDB + kb), *(const v8h*)(bBase + n * 16 * LDB + kb + 16), 0, 1, 2, 3, 4, 5, 6, 7, 8, 9, 10, 11, 12, 13, 14, 15);
            acc[n] = __builtin_amdgcn_wmma_f32_16x16x32_f16(
                false, a, false, b, (short)0, acc[n], false, false);
        }
    }

    const long rowb = mbase + hi * 8;
    float mk[8];
#pragma unroll
    for (int r = 0; r < 8; ++r) mk[r] = mask[rowb + r] ? 1.0f : 0.0f;

#pragma unroll
    for (int n = 0; n < 16; ++n) {
        const int col = n * 16 + l15;
        const float bb = bias[col];
#pragma unroll
        for (int r = 0; r < 8; ++r) {
            out[(size_t)(rowb + r) * HID + col] = (acc[n][r] + bb) * mk[r];
        }
    }
}

// ---------------------------------------------------------------------------
// Launch
// ---------------------------------------------------------------------------
extern "C" void kernel_launch(void* const* d_in, const int* in_sizes, int n_in,
                              void* d_out, int out_size, void* d_ws, size_t ws_size,
                              hipStream_t stream) {
    const float* bbox = (const float*)d_in[0];
    const int*   mask = (const int*)d_in[1];
    const float* W1   = (const float*)d_in[2];
    const float* b1   = (const float*)d_in[3];
    const float* g1   = (const float*)d_in[4];
    const float* be1  = (const float*)d_in[5];
    const float* W2   = (const float*)d_in[6];
    const float* b2   = (const float*)d_in[7];
    const float* g2   = (const float*)d_in[8];
    const float* be2  = (const float*)d_in[9];
    const float* W3   = (const float*)d_in[10];
    const float* b3   = (const float*)d_in[11];
    float* out = (float*)d_out;

    char* ws = (char*)d_ws;
    const size_t bytesH = (size_t)ROWS * HID * sizeof(_Float16);   // 134 MB each
    _Float16* H1  = (_Float16*)(ws);
    _Float16* H2  = (_Float16*)(ws + bytesH);
    _Float16* W2T = (_Float16*)(ws + 2 * bytesH);
    _Float16* W3T = (_Float16*)(ws + 2 * bytesH + (size_t)HID * HID * sizeof(_Float16));
    float* stats  = (float*)(ws + 2 * bytesH + 2 * (size_t)HID * HID * sizeof(_Float16));
    float* sum1 = stats;        float* sq1 = stats + 256;
    float* sc1  = stats + 512;  float* sh1 = stats + 768;
    float* sum2 = stats + 1024; float* sq2 = stats + 1280;
    float* sc2  = stats + 1536; float* sh2 = stats + 1792;
    float* cnt  = stats + 2048;

    const size_t ldsB = (size_t)HID * LDB * sizeof(_Float16);      // 132 KB

    k_zero_stats<<<9, 256, 0, stream>>>(stats, 2049);

    k_prep_weight<<<HID, HID, 0, stream>>>(W2, W2T);
    k_prep_weight<<<HID, HID, 0, stream>>>(W3, W3T);

    k1_layer1<<<ROWS / 32, 256, 0, stream>>>(bbox, mask, W1, b1, H1, sum1, sq1, cnt);

    k_finalize_bn<<<1, HID, 0, stream>>>(sum1, sq1, g1, be1, cnt, sc1, sh1);

    k3_gemm_bn_stats<<<ROWS / 128, 256, ldsB, stream>>>(H1, W2T, sc1, sh1, b2, mask,
                                                        H2, sum2, sq2);

    k_finalize_bn<<<1, HID, 0, stream>>>(sum2, sq2, g2, be2, cnt, sc2, sh2);

    k5_gemm_out<<<ROWS / 128, 256, ldsB, stream>>>(H2, W3T, sc2, sh2, b3, mask, out);
}
